// DoubleMHAttention_12103217840332
// MI455X (gfx1250) — matrix operand order repaired and checked by reading, not compiled, not run on hardware
//
#include <hip/hip_runtime.h>
#include <hip/hip_bf16.h>

// Problem constants (match reference)
constexpr int NN = 1024;   // sequence length
constexpr int BB = 4;      // batch
constexpr int CC = 1024;   // channels
constexpr int HH = 8;      // heads
constexpr int DH = 128;    // head dim
constexpr int NB = NN * BB;            // 4096 rows for projections
constexpr size_t PROJ = (size_t)BB * HH * NN * DH;  // elements per projection

typedef __attribute__((ext_vector_type(16))) _Float16 v16h;
typedef __attribute__((ext_vector_type(8)))  float    v8f;

// ---- WMMA fragment loaders (contiguous k = 16*half + i, same in A and B) -----------------
// A-matrix 16x32 f16: lane (m=lane&15, half=lane>>4) holds
//   halves[0..7]  = A[m, kbase + 16*half + i]
//   halves[8..15] = A[m, kbase + 16*half + 8 + i]
__device__ __forceinline__ v16h load_a_f32(const float* __restrict__ row, int kbase, int half) {
    v16h a;
    const float* p0 = row + kbase + 16 * half;
    const float* p1 = p0 + 8;
#pragma unroll
    for (int i = 0; i < 8; ++i) { a[i] = (_Float16)p0[i]; a[8 + i] = (_Float16)p1[i]; }
    return a;
}
__device__ __forceinline__ v16h load_a_f16(const _Float16* __restrict__ row, int kbase, int half) {
    v16h a;
    const _Float16* p0 = row + kbase + 16 * half;
    const _Float16* p1 = p0 + 8;
#pragma unroll
    for (int i = 0; i < 8; ++i) { a[i] = p0[i]; a[8 + i] = p1[i]; }
    return a;
}
// B-matrix 32x16 f16: lane (n=lane&15, half=lane>>4) holds
//   halves[i] = B[kbase + 16*half + i, n]   (pointer passed = data of
//   "column n", contiguous along k, i.e. a row of B^T)
__device__ __forceinline__ v16h load_b_f32(const float* __restrict__ col, int kbase, int half) {
    v16h b;
    const float* p = col + kbase + 16 * half;
#pragma unroll
    for (int i = 0; i < 16; ++i) b[i] = (_Float16)p[i];
    return b;
}
__device__ __forceinline__ v16h load_b_f16(const _Float16* __restrict__ col, int kbase, int half) {
    v16h b;
    const _Float16* p = col + kbase + 16 * half;
#pragma unroll
    for (int i = 0; i < 16; ++i) b[i] = p[i];
    return b;
}

__device__ __forceinline__ v8f wmma_f16(v16h a, v16h b, v8f c) {
    return __builtin_amdgcn_wmma_f32_16x16x32_f16(false, a, false, b, (short)0, c, false, false);
}

// ---------------------------------------------------------------------------
// Kernel 1: projection  Y = X @ W^T  (NT GEMM), f32 in -> f16 out
// One wave computes a 16-row x 128-col (one head) strip, so it owns every
// column of each of its 16 rows:
//   vmode==0 (q/k): L2-normalize rows in-register (shfl_xor butterfly leaves
//                   the row sum-of-squares in all lanes) and store f16 (B,H,N,D)
//   vmode==1 (v):   store transposed (B,H,D,N) so AV B-frags are contiguous
// ---------------------------------------------------------------------------
__global__ __launch_bounds__(32) void proj_kernel(
    const float* __restrict__ X, const float* __restrict__ W,
    _Float16* __restrict__ dst, int vmode) {
    const int blk = blockIdx.x;
    const int rt = blk >> 3;          // row tile over NB (0..255)
    const int hb = blk & 7;           // head block (0..7)
    const int lane = threadIdx.x & 31;
    const int lm = lane & 15, half = lane >> 4;

    const float* xrow = X + (size_t)(rt * 16 + lm) * CC;
    const int colbase = hb * DH;

    v8f acc[8] = {};
    for (int k = 0; k < CC; k += 32) {
        __builtin_prefetch(xrow + k + 64, 0, 1);   // global_prefetch_b8
        v16h a = load_a_f32(xrow, k, half);
#pragma unroll
        for (int t = 0; t < 8; ++t) {
            const float* wcol = W + (size_t)(colbase + t * 16 + lm) * CC;
            v16h bf = load_b_f32(wcol, k, half);
            acc[t] = wmma_f16(a, bf, acc[t]);
        }
    }

    // C/D layout: lane holds column (t*16 + lm), rows j + 8*half
    float rinv[8];
    if (!vmode) {
#pragma unroll
        for (int j = 0; j < 8; ++j) {
            float s = 0.f;
#pragma unroll
            for (int t = 0; t < 8; ++t) { const float y = acc[t][j]; s += y * y; }
#pragma unroll
            for (int off = 1; off < 16; off <<= 1) s += __shfl_xor(s, off, 32);
            rinv[j] = rsqrtf(s + 1e-30f);          // all lanes hold row sum
        }
    } else {
#pragma unroll
        for (int j = 0; j < 8; ++j) rinv[j] = 1.f;
    }

#pragma unroll
    for (int t = 0; t < 8; ++t) {
        const int dl = t * 16 + lm;
#pragma unroll
        for (int j = 0; j < 8; ++j) {
            const float y = acc[t][j] * rinv[j];
            const int r = rt * 16 + j + 8 * half;  // row in NB = n*B + b
            const int n = r >> 2;
            const int b = r & 3;
            const size_t bh = (size_t)(b * HH + hb);
            if (!vmode) dst[(bh * NN + n) * DH + dl] = (_Float16)y;
            else        dst[(bh * DH + dl) * NN + n] = (_Float16)y;
        }
    }
}

// ---------------------------------------------------------------------------
// Kernel 2: scores + dual softmax + combine.  One wave per (b,h, 16-row tile).
// q,k are unit-normalized -> every logit is a cosine similarity in [-1,1], so
// softmax needs NO max-subtraction: exp(s) in [1/e, e], row sums <= ~2800.
// Two recompute passes (sum -> write); no LDS, barrier-free.
// ---------------------------------------------------------------------------
__global__ __launch_bounds__(32) void attn_kernel(
    const _Float16* __restrict__ qr, const _Float16* __restrict__ kr,
    const _Float16* __restrict__ qc, const _Float16* __restrict__ kc,
    float* __restrict__ attn_out) {
    const int blk = blockIdx.x;
    const int rt = blk & 63;          // row tile (0..63)
    const int bh = blk >> 6;          // b*H + h  (0..31)
    const int lane = threadIdx.x & 31;
    const int lm = lane & 15, half = lane >> 4;

    const _Float16* krb = kr + (size_t)bh * NN * DH;
    const _Float16* kcb = kc + (size_t)bh * NN * DH;

    // Preload q A-fragments (D=128 -> 4 k-steps), both matrices
    v16h aR[4], aC[4];
    {
        const _Float16* q0 = qr + ((size_t)bh * NN + rt * 16 + lm) * DH;
        const _Float16* q1 = qc + ((size_t)bh * NN + rt * 16 + lm) * DH;
#pragma unroll
        for (int ks = 0; ks < 4; ++ks) {
            aR[ks] = load_a_f16(q0, ks * 32, half);
            aC[ks] = load_a_f16(q1, ks * 32, half);
        }
    }

    float dR[8], dC[8];
#pragma unroll
    for (int j = 0; j < 8; ++j) { dR[j] = 0.f; dC[j] = 0.f; }

    // ---- PASS 1: softmax denominators ----
    for (int ct = 0; ct < 64; ++ct) {
        const _Float16* krow = krb + (size_t)(ct * 16 + lm) * DH;
        const _Float16* crow = kcb + (size_t)(ct * 16 + lm) * DH;
        v8f sR = {}, sC = {};
#pragma unroll
        for (int ks = 0; ks < 4; ++ks) {
            sR = wmma_f16(aR[ks], load_b_f16(krow, ks * 32, half), sR);
            sC = wmma_f16(aC[ks], load_b_f16(crow, ks * 32, half), sC);
        }
#pragma unroll
        for (int j = 0; j < 8; ++j) {
            dR[j] += __expf(sR[j]);
            dC[j] += __expf(sC[j]);
        }
    }
#pragma unroll
    for (int j = 0; j < 8; ++j) {
#pragma unroll
        for (int off = 1; off < 16; off <<= 1) {
            dR[j] += __shfl_xor(dR[j], off, 32);
            dC[j] += __shfl_xor(dC[j], off, 32);
        }
        dR[j] = 0.5f / dR[j];          // fold in the 0.5 combine weight
        dC[j] = 0.5f / dC[j];
    }

    // ---- PASS 2: write combined probabilities ----
    float* out = attn_out + (size_t)bh * NN * NN;
    for (int ct = 0; ct < 64; ++ct) {
        const _Float16* krow = krb + (size_t)(ct * 16 + lm) * DH;
        const _Float16* crow = kcb + (size_t)(ct * 16 + lm) * DH;
        v8f sR = {}, sC = {};
#pragma unroll
        for (int ks = 0; ks < 4; ++ks) {
            sR = wmma_f16(aR[ks], load_b_f16(krow, ks * 32, half), sR);
            sC = wmma_f16(aC[ks], load_b_f16(crow, ks * 32, half), sC);
        }
#pragma unroll
        for (int j = 0; j < 8; ++j) {
            const float p = __expf(sR[j]) * dR[j] + __expf(sC[j]) * dC[j];
            const int row = rt * 16 + j + 8 * half;
            out[(size_t)row * NN + ct * 16 + lm] = p;
        }
    }
}

// ---------------------------------------------------------------------------
// Kernel 3: O = attn @ V.  One wave per (b,h, 16-row tile) computes 16x128.
// attn read back as f32 -> f16 A-frags; V^T gives contiguous B-frags.
// Output written in (N, B, C) layout.
// ---------------------------------------------------------------------------
__global__ __launch_bounds__(32) void av_kernel(
    const float* __restrict__ attn, const _Float16* __restrict__ vT,
    float* __restrict__ xout) {
    const int blk = blockIdx.x;
    const int rt = blk & 63;
    const int bh = blk >> 6;
    const int h = bh & 7;
    const int b = bh >> 3;
    const int lane = threadIdx.x & 31;
    const int lm = lane & 15, half = lane >> 4;

    const float* arow = attn + ((size_t)bh * NN + rt * 16 + lm) * NN;
    const _Float16* vb = vT + (size_t)bh * DH * NN;

    v8f acc[8] = {};
    for (int k = 0; k < NN; k += 32) {
        __builtin_prefetch(arow + k + 64, 0, 1);
        v16h a = load_a_f32(arow, k, half);
#pragma unroll
        for (int t = 0; t < 8; ++t) {
            const _Float16* vcol = vb + (size_t)(t * 16 + lm) * NN;
            acc[t] = wmma_f16(a, load_b_f16(vcol, k, half), acc[t]);
        }
    }
#pragma unroll
    for (int t = 0; t < 8; ++t) {
        const int dl = t * 16 + lm;
#pragma unroll
        for (int j = 0; j < 8; ++j) {
            const int n = rt * 16 + j + 8 * half;
            xout[((size_t)n * BB + b) * CC + h * DH + dl] = acc[t][j];
        }
    }
}

// ---------------------------------------------------------------------------
extern "C" void kernel_launch(void* const* d_in, const int* in_sizes, int n_in,
                              void* d_out, int out_size, void* d_ws, size_t ws_size,
                              hipStream_t stream) {
    (void)in_sizes; (void)n_in; (void)out_size; (void)ws_size;
    const float* x_qr = (const float*)d_in[0];
    const float* x_kr = (const float*)d_in[1];
    const float* x_qc = (const float*)d_in[2];
    const float* x_kc = (const float*)d_in[3];
    const float* x_v  = (const float*)d_in[4];
    const float* W_qr = (const float*)d_in[5];
    const float* W_kr = (const float*)d_in[6];
    const float* W_qc = (const float*)d_in[7];
    const float* W_kc = (const float*)d_in[8];
    const float* W_v  = (const float*)d_in[9];

    float* xout = (float*)d_out;                          // (N,B,C)
    float* attn = xout + (size_t)NN * BB * CC;            // (B,H,N,N)

    _Float16* ws16 = (_Float16*)d_ws;
    _Float16* qr = ws16 + 0 * PROJ;                       // normalized (B,H,N,D)
    _Float16* kr = ws16 + 1 * PROJ;
    _Float16* qc = ws16 + 2 * PROJ;
    _Float16* kc = ws16 + 3 * PROJ;
    _Float16* vT = ws16 + 4 * PROJ;                       // (B,H,D,N)

    const dim3 blk(32);
    const int projBlocks = (NB / 16) * (CC / DH);         // 256 * 8 = 2048
    proj_kernel<<<projBlocks, blk, 0, stream>>>(x_qr, W_qr, qr, 0);
    proj_kernel<<<projBlocks, blk, 0, stream>>>(x_kr, W_kr, kr, 0);
    proj_kernel<<<projBlocks, blk, 0, stream>>>(x_qc, W_qc, qc, 0);
    proj_kernel<<<projBlocks, blk, 0, stream>>>(x_kc, W_kc, kc, 0);
    proj_kernel<<<projBlocks, blk, 0, stream>>>(x_v,  W_v,  vT, 1);

    const int attnBlocks = BB * HH * (NN / 16);           // 32 * 64 = 2048
    attn_kernel<<<attnBlocks, blk, 0, stream>>>(qr, kr, qc, kc, attn);
    av_kernel<<<attnBlocks, blk, 0, stream>>>(attn, vT, xout);
}
